// HstuBlock_35304631173849
// MI455X (gfx1250) — matrix operand order repaired and checked by reading, not compiled, not run on hardware
//
#include <hip/hip_runtime.h>
#include <cstdint>
#include <cstddef>

// ---------------------------------------------------------------------------
// HSTU block for MI455X (gfx1250): bf16 WMMA everywhere, flash-style fused
// attention, block-level LDS staging of K/V (async-to-LDS when available),
// LDS re-layout for the S-tile. Wave tile = 32 rows x 64 cols.
// ---------------------------------------------------------------------------

typedef __attribute__((ext_vector_type(16))) __bf16 v16bf;
typedef __attribute__((ext_vector_type(8)))  float  v8f;
typedef int v4i_g __attribute__((vector_size(16)));   // int4, GCC vector style

union V16BF { v16bf v; unsigned short u[16]; };
union V8F   { v8f  v; float f[8]; };

#define DEV static __device__ __forceinline__
#define AS1 __attribute__((address_space(1)))
#define AS3 __attribute__((address_space(3)))

#if defined(__has_builtin)
#  if __has_builtin(__builtin_amdgcn_global_load_async_to_lds_b128)
#    define HAVE_ASYNC_LDS 1
#  endif
#endif
#ifndef HAVE_ASYNC_LDS
#  define HAVE_ASYNC_LDS 0
#endif

constexpr int BB   = 8;
constexpr int NN   = 2048;
constexpr int EMB  = 512;
constexpr int ATT  = 64;
constexpr int H    = 8;
constexpr int ROWS = BB * NN;     // 16384
constexpr int PROJ = 2048;        // LIN*2*H + ATT*H*2
constexpr float EPSV = 1e-6f;

DEV unsigned short f2bf(float x) {                 // round-to-nearest-even
  unsigned int u = __builtin_bit_cast(unsigned int, x);
  u = (u + 0x7FFFu + ((u >> 16) & 1u)) >> 16;
  return (unsigned short)u;
}

// fast silu: v_exp_f32 + v_rcp_f32 (no IEEE divide sequence)
DEV float siluf(float x) {
  return x * __builtin_amdgcn_rcpf(1.0f + __expf(-x));
}

DEV v8f v8zero() {
  V8F z;
  #pragma unroll
  for (int i = 0; i < 8; ++i) z.f[i] = 0.0f;
  return z.v;
}

DEV v8f wmma_bf16(v16bf a, v16bf b, v8f c) {
  // D = A(16x32) * B(32x16) + C, f32 accum  ->  v_wmma_f32_16x16x32_bf16
  return __builtin_amdgcn_wmma_f32_16x16x32_bf16(false, a, false, b,
                                                 (short)0, c, false, false);
}

DEV void wait_asynccnt0() {
#if defined(__has_builtin)
#  if __has_builtin(__builtin_amdgcn_s_wait_asynccnt)
  __builtin_amdgcn_s_wait_asynccnt(0);
  return;
#  endif
#endif
  asm volatile("s_wait_asynccnt 0" ::: "memory");
}

// --- Fragment loaders (wave32, contiguous k order shared by A and B) ---------------------------
// A 16x32 bf16: lane L -> row = L&15 ; element e -> k = e+(L>>4)*16
DEV v16bf load_A(const unsigned short* p, int ld, int row0, int k0, int lane) {
  V16BF a;
  const int r  = row0 + (lane & 15);
  const int hb = (lane >> 4) * 16;
  const unsigned short* pr = p + (size_t)r * ld + k0;
  #pragma unroll
  for (int e = 0; e < 16; ++e) {
    const int kk = e + hb;
    a.u[e] = pr[kk];
  }
  return a.v;
}

// B 32x16 bf16 from [col][k] memory (k contiguous): lane L -> col = L&15,
// k = e + (L>>4)*16 -> one contiguous 32B run per lane.
DEV v16bf load_BT(const unsigned short* p, int ld, int k0, int col0, int lane) {
  V16BF b;
  const int c = col0 + (lane & 15);
  const unsigned short* pc = p + (size_t)c * ld + k0 + ((lane >> 4) << 4);
  #pragma unroll
  for (int e = 0; e < 16; ++e) b.u[e] = pc[e];
  return b.v;
}

// B 32x16 bf16 from [k][col] memory (k strided) — used for the v operand.
DEV v16bf load_BS(const unsigned short* p, int ld, int k0, int col0, int lane) {
  V16BF b;
  const int c = col0 + (lane & 15);
  const unsigned short* pk = p + (size_t)(k0 + ((lane >> 4) << 4)) * ld + c;
  #pragma unroll
  for (int e = 0; e < 16; ++e) b.u[e] = pk[(size_t)e * ld];
  return b.v;
}

// --- Kernel 0a/0b: weight packing to bf16 -----------------------------------
__global__ void pack_uvqkT_kernel(const float* __restrict__ uvqk,
                                  unsigned short* __restrict__ dst) {
  const int idx = blockIdx.x * blockDim.x + threadIdx.x;   // PROJ*EMB
  const int c = idx >> 9;            // output column of proj
  const int d = idx & 511;           // reduction dim
  dst[(size_t)c * EMB + d] = f2bf(uvqk[(size_t)d * PROJ + c]);
}

__global__ void pack_ow_kernel(const float* __restrict__ ow,
                               unsigned short* __restrict__ dst) {
  const int idx = blockIdx.x * blockDim.x + threadIdx.x;   // EMB*EMB, [e][f]
  dst[idx] = f2bf(ow[idx]);
}

// --- Kernel 1: LayerNorm rows -> bf16 ---------------------------------------
__global__ void ln_kernel(const float* __restrict__ x,
                          unsigned short* __restrict__ normed) {
  __shared__ float s1[256];
  __shared__ float s2[256];
  const int tid = threadIdx.x;
  const size_t base = (size_t)blockIdx.x * EMB;
  const float a = x[base + tid];
  const float b = x[base + tid + 256];
  s1[tid] = a + b;
  s2[tid] = a * a + b * b;
  __syncthreads();
  #pragma unroll
  for (int off = 128; off > 0; off >>= 1) {
    if (tid < off) { s1[tid] += s1[tid + off]; s2[tid] += s2[tid + off]; }
    __syncthreads();
  }
  const float mean = s1[0] * (1.0f / EMB);
  const float var  = s2[0] * (1.0f / EMB) - mean * mean;
  const float sc   = rsqrtf(var + EPSV);
  normed[base + tid]       = f2bf((a - mean) * sc);
  normed[base + tid + 256] = f2bf((b - mean) * sc);
}

// --- Kernel 2: proj = silu(normed @ uvqk), scatter u/v/q/k ------------------
// wave = 32 rows x 64 cols; B fragments reused across both 16-row tiles.
__global__ void proj_gemm_kernel(const unsigned short* __restrict__ normed,
                                 const unsigned short* __restrict__ uvqkT,
                                 float* __restrict__ u_out,
                                 unsigned short* __restrict__ v_out,
                                 unsigned short* __restrict__ q_out,
                                 unsigned short* __restrict__ k_out) {
  const int wid  = __builtin_amdgcn_readfirstlane(threadIdx.x >> 5); // uniform
  const int lane = threadIdx.x & 31;
  const int gw = blockIdx.x * 8 + wid;
  const int CG = PROJ / 64;                       // 32 column groups
  const int rowTile = gw / CG, cg = gw % CG;
  const int row0 = rowTile * 32, col0 = cg * 64;

  v8f acc[2][4];
  #pragma unroll
  for (int i = 0; i < 2; ++i)
    #pragma unroll
    for (int j = 0; j < 4; ++j) acc[i][j] = v8zero();

  for (int k0 = 0; k0 < EMB; k0 += 32) {
    v16bf a[2];
    a[0] = load_A(normed, EMB, row0,      k0, lane);
    a[1] = load_A(normed, EMB, row0 + 16, k0, lane);
    #pragma unroll
    for (int j = 0; j < 4; ++j) {
      const v16bf b = load_BT(uvqkT, EMB, k0, col0 + 16 * j, lane);
      acc[0][j] = wmma_bf16(a[0], b, acc[0][j]);
      acc[1][j] = wmma_bf16(a[1], b, acc[1][j]);
    }
  }
  const int half = lane >> 4, cl = lane & 15;
  #pragma unroll
  for (int j = 0; j < 4; ++j) {
    const int colbase = col0 + 16 * j;
    const int seg = colbase >> 9;                  // wave-uniform (SGPR)
    const int col = colbase + cl;
    #pragma unroll
    for (int i = 0; i < 2; ++i) {
      V8F d; d.v = acc[i][j];
      #pragma unroll
      for (int r = 0; r < 8; ++r) {
        const int row = row0 + i * 16 + r + half * 8;
        const float s = siluf(d.f[r]);
        const size_t ri = (size_t)row * EMB;
        if (seg == 0)       u_out[ri + col]          = s;
        else if (seg == 1)  v_out[ri + (col - 512)]  = f2bf(s);
        else if (seg == 2)  q_out[ri + (col - 1024)] = f2bf(s);
        else                k_out[ri + (col - 1536)] = f2bf(s);
      }
    }
  }
}

// --- Kernel 3: fused attention (no qk materialization) ----------------------
// block = 8 waves sharing one (b,h); wave = 32 query rows.
// Per 32-key chunk: K/V tiles staged once into LDS (async if available) and
// consumed by all 8 waves:  S = q.k^T ; S = silu(S+bias)*mask ; attn += S.v
__global__ void attn_kernel(const unsigned short* __restrict__ q,
                            const unsigned short* __restrict__ k,
                            const unsigned short* __restrict__ v,
                            const float* __restrict__ bias,
                            const float* __restrict__ mask,
                            float* __restrict__ attn) {
  __shared__ unsigned short kTile[32 * 64];       // [key][d] for this chunk
  __shared__ unsigned short vTile[32 * 64];       // [key][d] for this chunk
  __shared__ unsigned short sS[8][32 * 32];       // per-wave S tile (bf16)

  const int wid  = __builtin_amdgcn_readfirstlane(threadIdx.x >> 5); // uniform
  const int lane = threadIdx.x & 31;
  const int tid  = threadIdx.x;

  // block -> (b, h, group of 8 query tiles); 8 waves share (b,h) => shared K/V
  const int gb     = blockIdx.x;                  // 0..511
  const int tgroup = gb & 7;
  const int h      = (gb >> 3) & 7;
  const int b      = gb >> 6;
  const int ntile  = tgroup * 8 + wid;
  const int row0   = ntile * 32;

  const unsigned short* qp = q + (size_t)b * NN * EMB + h * ATT;
  const unsigned short* kp = k + (size_t)b * NN * EMB + h * ATT;
  const unsigned short* vp = v + (size_t)b * NN * EMB + h * ATT;
  const float* biasp = bias + (size_t)h * NN * NN;
  const float* maskp = mask + (size_t)b * NN * NN;

  v16bf aq[2][2];
  #pragma unroll
  for (int i = 0; i < 2; ++i) {
    aq[i][0] = load_A(qp, EMB, row0 + 16 * i, 0, lane);
    aq[i][1] = load_A(qp, EMB, row0 + 16 * i, 32, lane);
  }
  v8f acc[2][4];
  #pragma unroll
  for (int i = 0; i < 2; ++i)
    #pragma unroll
    for (int j = 0; j < 4; ++j) acc[i][j] = v8zero();
  const v8f zero = v8zero();
  const int half = lane >> 4, cl = lane & 15;
  unsigned short* sw = &sS[wid][0];

  // cooperative staging slot: 256 threads x 16B covers one 4KB tile
  const int skey = tid >> 3;                      // 0..31
  const int sd   = (tid & 7) * 8;                 // 0..56, 8 bf16 = 16B

  for (int m0 = 0; m0 < NN; m0 += 32) {
    __syncthreads();                              // prev chunk fully consumed
#if HAVE_ASYNC_LDS
    __builtin_amdgcn_global_load_async_to_lds_b128(
        (AS1 v4i_g*)(kp + (size_t)(m0 + skey) * EMB + sd),
        (AS3 v4i_g*)&kTile[skey * 64 + sd], 0, 0);
    __builtin_amdgcn_global_load_async_to_lds_b128(
        (AS1 v4i_g*)(vp + (size_t)(m0 + skey) * EMB + sd),
        (AS3 v4i_g*)&vTile[skey * 64 + sd], 0, 0);
    wait_asynccnt0();
#else
    *(uint4*)&kTile[skey * 64 + sd] =
        *(const uint4*)(kp + (size_t)(m0 + skey) * EMB + sd);
    *(uint4*)&vTile[skey * 64 + sd] =
        *(const uint4*)(vp + (size_t)(m0 + skey) * EMB + sd);
#endif
    __syncthreads();                              // tiles ready for all waves

    // prefetch next chunk's bias/mask lines into L2 (speculative, OOB dropped)
    __builtin_prefetch(biasp + (size_t)(row0 + cl) * NN + m0 + 32, 0, 0);
    __builtin_prefetch(maskp + (size_t)(row0 + cl) * NN + m0 + 32, 0, 0);

    #pragma unroll
    for (int j = 0; j < 2; ++j) {
      const v16bf bk0 = load_BT(kTile, 64, 0,  16 * j, lane);   // ds reads
      const v16bf bk1 = load_BT(kTile, 64, 32, 16 * j, lane);
      const int col = m0 + 16 * j + cl;
      #pragma unroll
      for (int i = 0; i < 2; ++i) {
        v8f s = wmma_bf16(aq[i][0], bk0, zero);
        s = wmma_bf16(aq[i][1], bk1, s);
        V8F sd2; sd2.v = s;
        #pragma unroll
        for (int r = 0; r < 8; ++r) {
          const int n = row0 + 16 * i + r + half * 8;
          float t = sd2.f[r] + biasp[(size_t)n * NN + col];
          t = siluf(t) * maskp[(size_t)n * NN + col];
          sw[(16 * i + r + half * 8) * 32 + (16 * j + cl)] = f2bf(t);
        }
      }
    }
    // re-layout S (C/D frag -> A frag) through LDS (same-wave ds in-order)
    v16bf aS[2];
    aS[0] = load_A(sw, 32, 0,  0, lane);
    aS[1] = load_A(sw, 32, 16, 0, lane);
    #pragma unroll
    for (int j2 = 0; j2 < 4; ++j2) {
      const v16bf bv = load_BS(vTile, 64, 0, 16 * j2, lane);    // ds reads
      acc[0][j2] = wmma_bf16(aS[0], bv, acc[0][j2]);
      acc[1][j2] = wmma_bf16(aS[1], bv, acc[1][j2]);
    }
  }

  #pragma unroll
  for (int i = 0; i < 2; ++i)
    #pragma unroll
    for (int j2 = 0; j2 < 4; ++j2) {
      V8F d; d.v = acc[i][j2];
      const int col = h * ATT + 16 * j2 + cl;
      #pragma unroll
      for (int r = 0; r < 8; ++r) {
        const int n = row0 + 16 * i + r + half * 8;
        attn[((size_t)b * NN + n) * EMB + col] = d.f[r];
      }
    }
}

// --- Kernel 4: gated = u * layer_norm(attn) -> bf16 -------------------------
__global__ void gated_kernel(const float* __restrict__ attn,
                             const float* __restrict__ u,
                             unsigned short* __restrict__ gated) {
  __shared__ float s1[256];
  __shared__ float s2[256];
  const int tid = threadIdx.x;
  const size_t base = (size_t)blockIdx.x * EMB;
  const float a = attn[base + tid];
  const float b = attn[base + tid + 256];
  s1[tid] = a + b;
  s2[tid] = a * a + b * b;
  __syncthreads();
  #pragma unroll
  for (int off = 128; off > 0; off >>= 1) {
    if (tid < off) { s1[tid] += s1[tid + off]; s2[tid] += s2[tid + off]; }
    __syncthreads();
  }
  const float mean = s1[0] * (1.0f / EMB);
  const float var  = s2[0] * (1.0f / EMB) - mean * mean;
  const float sc   = rsqrtf(var + EPSV);
  gated[base + tid]       = f2bf(u[base + tid]       * (a - mean) * sc);
  gated[base + tid + 256] = f2bf(u[base + tid + 256] * (b - mean) * sc);
}

// --- Kernel 5: out = x + gated @ o_w^T + o_b --------------------------------
__global__ void out_gemm_kernel(const unsigned short* __restrict__ gated,
                                const unsigned short* __restrict__ owb,
                                const float* __restrict__ ob,
                                const float* __restrict__ x,
                                float* __restrict__ out) {
  const int wid  = __builtin_amdgcn_readfirstlane(threadIdx.x >> 5); // uniform
  const int lane = threadIdx.x & 31;
  const int gw = blockIdx.x * 8 + wid;
  const int CG = EMB / 64;                         // 8 column groups
  const int rowTile = gw / CG, cg = gw % CG;
  const int row0 = rowTile * 32, col0 = cg * 64;

  v8f acc[2][4];
  #pragma unroll
  for (int i = 0; i < 2; ++i)
    #pragma unroll
    for (int j = 0; j < 4; ++j) acc[i][j] = v8zero();

  for (int k0 = 0; k0 < EMB; k0 += 32) {
    v16bf a[2];
    a[0] = load_A(gated, EMB, row0,      k0, lane);
    a[1] = load_A(gated, EMB, row0 + 16, k0, lane);
    #pragma unroll
    for (int j = 0; j < 4; ++j) {
      // o_w is [e][f] with f contiguous == exactly the [col][k] layout we want
      const v16bf b = load_BT(owb, EMB, k0, col0 + 16 * j, lane);
      acc[0][j] = wmma_bf16(a[0], b, acc[0][j]);
      acc[1][j] = wmma_bf16(a[1], b, acc[1][j]);
    }
  }
  const int half = lane >> 4, cl = lane & 15;
  #pragma unroll
  for (int j = 0; j < 4; ++j) {
    const int col = col0 + 16 * j + cl;
    const float bcol = ob[col];
    #pragma unroll
    for (int i = 0; i < 2; ++i) {
      V8F d; d.v = acc[i][j];
      #pragma unroll
      for (int r = 0; r < 8; ++r) {
        const int row = row0 + i * 16 + r + half * 8;
        const size_t idx = (size_t)row * EMB + col;
        out[idx] = x[idx] + d.f[r] + bcol;
      }
    }
  }
}

// ---------------------------------------------------------------------------
extern "C" void kernel_launch(void* const* d_in, const int* in_sizes, int n_in,
                              void* d_out, int out_size, void* d_ws, size_t ws_size,
                              hipStream_t stream) {
  (void)in_sizes; (void)n_in; (void)out_size; (void)ws_size;
  const float* x    = (const float*)d_in[0];
  const float* bias = (const float*)d_in[1];
  const float* mask = (const float*)d_in[2];
  const float* uvqk = (const float*)d_in[3];
  const float* ow   = (const float*)d_in[4];
  const float* ob   = (const float*)d_in[5];
  float* out = (float*)d_out;

  char* w = (char*)d_ws;
  unsigned short* uvqkT  = (unsigned short*)w; w += (size_t)PROJ * EMB * 2;  //   2 MB
  unsigned short* owb    = (unsigned short*)w; w += (size_t)EMB * EMB * 2;   // 0.5 MB
  unsigned short* normed = (unsigned short*)w; w += (size_t)ROWS * EMB * 2;  //  16 MB
  float*          u_ws   = (float*)w;          w += (size_t)ROWS * EMB * 4;  //  32 MB
  unsigned short* v_ws   = (unsigned short*)w; w += (size_t)ROWS * EMB * 2;  //  16 MB
  unsigned short* q_ws   = (unsigned short*)w; w += (size_t)ROWS * EMB * 2;  //  16 MB
  unsigned short* k_ws   = (unsigned short*)w; w += (size_t)ROWS * EMB * 2;  //  16 MB
  float*          attnws = (float*)w;          w += (size_t)ROWS * EMB * 4;  //  32 MB
  unsigned short* gated  = (unsigned short*)w;                               //  16 MB

  pack_uvqkT_kernel<<<(PROJ * EMB) / 256, 256, 0, stream>>>(uvqk, uvqkT);
  pack_ow_kernel<<<(EMB * EMB) / 256, 256, 0, stream>>>(ow, owb);
  ln_kernel<<<ROWS, 256, 0, stream>>>(x, normed);
  proj_gemm_kernel<<<(ROWS / 32) * (PROJ / 64) / 8, 256, 0, stream>>>(
      normed, uvqkT, u_ws, v_ws, q_ws, k_ws);
  attn_kernel<<<(BB * H * (NN / 32)) / 8, 256, 0, stream>>>(
      q_ws, k_ws, v_ws, bias, mask, attnws);
  gated_kernel<<<ROWS, 256, 0, stream>>>(attnws, u_ws, gated);
  out_gemm_kernel<<<(ROWS / 32) * (EMB / 64) / 8, 256, 0, stream>>>(
      gated, owb, ob, x, out);
}
